// VulnerabilityAwareAttention_33775622816232
// MI455X (gfx1250) — matrix operand order repaired and checked by reading, not compiled, not run on hardware
//
#include <hip/hip_runtime.h>
#include <hip/hip_bf16.h>
#include <math.h>

#define N 4096
#define D 256
#define H 4
#define HD 64
#define G 16
#define LN_EPS 1e-5f

typedef __attribute__((ext_vector_type(16))) _Float16 v16h;
typedef __attribute__((ext_vector_type(8)))  float    v8f;

union AFrag { v16h v; _Float16 h[16]; uint4 u[2]; };
union CFrag { v8f  v; float    f[8];  };

// ---- CDNA5 async global->LDS copy (ASYNCcnt) -------------------------------
__device__ __forceinline__ unsigned lds_u32(const void* p) {
  // generic pointer to LDS: addr[31:0] is the wave-relative LDS offset
  return (unsigned)(unsigned long long)p;
}
__device__ __forceinline__ void async_load_b128(unsigned lds_off, const void* gptr) {
  asm volatile("global_load_async_to_lds_b128 %0, %1, off"
               :: "v"(lds_off), "v"((unsigned long long)gptr) : "memory");
}
__device__ __forceinline__ void wait_asynccnt0() {
  asm volatile("s_wait_asynccnt 0x0" ::: "memory");
}

__global__ void f32_to_f16_kernel(const float* __restrict__ src, _Float16* __restrict__ dst, int n) {
  int i = blockIdx.x * 256 + threadIdx.x;
  if (i < n) dst[i] = (_Float16)src[i];
}

// transpose while converting: src is R x C row-major f32, dst is C x R row-major f16
__global__ void f32_to_f16_T_kernel(const float* __restrict__ src, _Float16* __restrict__ dst,
                                    int R, int C) {
  int i = blockIdx.x * 256 + threadIdx.x;
  if (i < R * C) {
    int r = i / C, c = i % C;
    dst[(size_t)c * R + r] = (_Float16)src[i];
  }
}

// gstart[g] = first row index with batch[i] >= g (batch sorted); gstart[G] = N
__global__ void find_gstart_kernel(const int* __restrict__ batch, int* __restrict__ gstart) {
  int g = threadIdx.x;
  if (g > G) return;
  int lo = 0, hi = N;
  while (lo < hi) { int mid = (lo + hi) >> 1; if (batch[mid] < g) lo = mid + 1; else hi = mid; }
  gstart[g] = lo;
}

// ---- WMMA GEMM: C[M x Nn] = A[M x K] @ Bt^T (+bias) (+rowAdd[batch[row]]) ---
// Bt is the N-major (transposed) weight: Bt[n][k], leading dim K.
// One wave computes a 16x64 tile. ACT: 0=none,1=relu,2=tanh.
// OUTH_MODE: 0=none, 1=row-major f16, 2=transposed f16 (outH[col*M+row]).
template <int ACT, bool ROWADD, bool OUTF, int OUTH_MODE>
__global__ void wmma_gemm_kernel(const _Float16* __restrict__ A, const _Float16* __restrict__ Bt,
                                 const float* __restrict__ bias,
                                 const float* __restrict__ rowAdd, const int* __restrict__ batch,
                                 float* __restrict__ outF, _Float16* __restrict__ outH,
                                 int M, int K, int Nn) {
  __shared__ __align__(16) _Float16 As[16][32];
  __shared__ __align__(16) _Float16 Bs[64][32];   // n-major: Bs[n][k]
  const int lane = threadIdx.x;
  const int half = lane >> 4, lm = lane & 15;
  const int i0 = blockIdx.x * 16;
  const int j0 = blockIdx.y * 64;

  CFrag acc[4];
#pragma unroll
  for (int t = 0; t < 4; t++)
#pragma unroll
    for (int r = 0; r < 8; r++) acc[t].f[r] = 0.0f;

  for (int kk = 0; kk < K; kk += 32) {
    // A tile 16x32: lane stages 16 halves of row lm
    {
      const _Float16* src = A + (size_t)(i0 + lm) * K + kk + half * 16;
      *(uint4*)&As[lm][half * 16]     = ((const uint4*)src)[0];
      *(uint4*)&As[lm][half * 16 + 8] = ((const uint4*)src)[1];
    }
    // B tile 64(n) x 32(k): lane stages rows j0+lane and j0+32+lane of Bt
    {
      const _Float16* s0 = Bt + (size_t)(j0 + lane) * K + kk;
      const _Float16* s1 = Bt + (size_t)(j0 + 32 + lane) * K + kk;
      uint4* d0 = (uint4*)&Bs[lane][0];
      uint4* d1 = (uint4*)&Bs[32 + lane][0];
#pragma unroll
      for (int q = 0; q < 4; q++) { d0[q] = ((const uint4*)s0)[q]; d1[q] = ((const uint4*)s1)[q]; }
    }
    __syncthreads();
    AFrag a;           // A frag: two contiguous 8-half runs
    a.u[0] = *(const uint4*)&As[lm][16 * half];
    a.u[1] = *(const uint4*)&As[lm][16 * half + 8];
#pragma unroll
    for (int t = 0; t < 4; t++) {
      AFrag b;         // B frag: contiguous along k in n-major storage
      b.u[0] = *(const uint4*)&Bs[t * 16 + lm][16 * half];
      b.u[1] = *(const uint4*)&Bs[t * 16 + lm][16 * half + 8];
      acc[t].v = __builtin_amdgcn_wmma_f32_16x16x32_f16(false, a.v, false, b.v,
                                                        (short)0, acc[t].v, false, false);
    }
    __syncthreads();
  }

#pragma unroll
  for (int t = 0; t < 4; t++)
#pragma unroll
    for (int r = 0; r < 8; r++) {
      int row = i0 + r + 8 * half;
      int col = j0 + t * 16 + lm;
      float v = acc[t].f[r] + bias[col];
      if constexpr (ROWADD) v += rowAdd[(size_t)batch[row] * Nn + col];
      if constexpr (ACT == 1) v = fmaxf(v, 0.0f);
      if constexpr (ACT == 2) v = tanhf(v);
      if constexpr (OUTF) outF[(size_t)row * Nn + col] = v;
      if constexpr (OUTH_MODE == 1) outH[(size_t)row * Nn + col] = (_Float16)v;
      if constexpr (OUTH_MODE == 2) outH[(size_t)col * M + row] = (_Float16)v;
    }
}

// per-graph max over x rows -> gctx[G][D]
__global__ void seg_max_x_kernel(const float* __restrict__ x, const int* __restrict__ gstart,
                                 float* __restrict__ gctx) {
  int g = blockIdx.x, d = threadIdx.x;
  int s = gstart[g], e = gstart[g + 1];
  float m = -__builtin_inff();
  for (int r = s; r < e; r++) m = fmaxf(m, x[(size_t)r * D + d]);
  gctx[(size_t)g * D + d] = m;
}

// gpart[g][j] = sum_k gctx[g][k] * Wc1[D + k][j]   (bottom half of Wc1)
__global__ void gpart_kernel(const float* __restrict__ gctx, const float* __restrict__ Wc1,
                             float* __restrict__ gpart) {
  int g = blockIdx.x, j = threadIdx.x;
  float s = 0.0f;
  for (int k = 0; k < D; k++) s += gctx[(size_t)g * D + k] * Wc1[(size_t)(D + k) * D + j];
  gpart[(size_t)g * D + j] = s;
}

// out[row] = sigmoid( dot(A[row,:L], w) + b )
__global__ void rowdot_sigmoid_kernel(const float* __restrict__ A, const float* __restrict__ w,
                                      const float* __restrict__ b, float* __restrict__ out, int L) {
  int row = blockIdx.x, lane = threadIdx.x;
  float s = 0.0f;
  for (int j = lane; j < L; j += 32) s += A[(size_t)row * L + j] * w[j];
#pragma unroll
  for (int m = 16; m >= 1; m >>= 1) s += __shfl_xor(s, m, 32);
  if (lane == 0) out[row] = 1.0f / (1.0f + __expf(-(s + b[0])));
}

// ---- Flash attention: one wave per (16-query tile, head) -------------------
// Qh/Kh row-major [N][D] f16; VhT transposed [D][N] f16.
__global__ void flash_attn_kernel(const _Float16* __restrict__ Qh, const _Float16* __restrict__ Kh,
                                  const _Float16* __restrict__ VhT, const int* __restrict__ batch,
                                  const int* __restrict__ gstart, float* __restrict__ attended) {
  __shared__ __align__(16) _Float16 Qs[16][64];
  __shared__ __align__(16) _Float16 Ks[32][64];   // Ks[key][kdim]
  __shared__ __align__(16) _Float16 Vt[64][32];   // Vt[dim][key]
  __shared__ __align__(16) _Float16 Ps[16][32];
  const int lane = threadIdx.x;
  const int half = lane >> 4, lm = lane & 15;
  const int q0 = blockIdx.x * 16;
  const int h  = blockIdx.y;
  const int colbase = h * HD;
  const float INF = __builtin_inff();
  const float scale = 0.125f;  // 1/sqrt(64)

  // Q tile (16x64)
  {
    const _Float16* src = Qh + (size_t)(q0 + lm) * D + colbase + half * 32;
    uint4* dst = (uint4*)&Qs[lm][half * 32];
#pragma unroll
    for (int q = 0; q < 4; q++) dst[q] = ((const uint4*)src)[q];
  }
  __syncthreads();
  AFrag aq[2];
#pragma unroll
  for (int c = 0; c < 2; c++) {
    aq[c].u[0] = *(const uint4*)&Qs[lm][c * 32 + 16 * half];
    aq[c].u[1] = *(const uint4*)&Qs[lm][c * 32 + 16 * half + 8];
  }

  float mrow[8], lrow[8];
  CFrag out[4];
#pragma unroll
  for (int r = 0; r < 8; r++) { mrow[r] = -INF; lrow[r] = 0.0f; }
#pragma unroll
  for (int t = 0; t < 4; t++)
#pragma unroll
    for (int r = 0; r < 8; r++) out[t].f[r] = 0.0f;

  int bq[8];
#pragma unroll
  for (int r = 0; r < 8; r++) bq[r] = batch[q0 + r + 8 * half];
  const int g_lo = batch[q0], g_hi = batch[q0 + 15];
  const int kstart = gstart[g_lo] & ~31;   // kb stays 32-aligned, so kb+31 < N always
  const int kend   = gstart[g_hi + 1];

  for (int kb = kstart; kb < kend; kb += 32) {
    // stage K rows (32x64) and transposed V (64x32) via async global->LDS DMA
    {
      const _Float16* ks = Kh + (size_t)(kb + lane) * D + colbase;
      unsigned kl = lds_u32(&Ks[lane][0]);
#pragma unroll
      for (int q = 0; q < 8; q++) async_load_b128(kl + q * 16, ks + q * 8);
      const _Float16* v0 = VhT + (size_t)(colbase + lane) * N + kb;
      const _Float16* v1 = VhT + (size_t)(colbase + 32 + lane) * N + kb;
      unsigned l0 = lds_u32(&Vt[lane][0]);
      unsigned l1 = lds_u32(&Vt[32 + lane][0]);
#pragma unroll
      for (int q = 0; q < 4; q++) {
        async_load_b128(l0 + q * 16, v0 + q * 8);
        async_load_b128(l1 + q * 16, v1 + q * 8);
      }
      if (kb + 32 < kend)
        __builtin_prefetch((const void*)(Kh + (size_t)(kb + 32 + lane) * D + colbase), 0, 1);
    }
    wait_asynccnt0();
    __syncthreads();

    // S = Q @ K^T : K^T columns are K rows -> contiguous fragment reads
    CFrag s[2];
#pragma unroll
    for (int c = 0; c < 2; c++) {
#pragma unroll
      for (int r = 0; r < 8; r++) s[c].f[r] = 0.0f;
#pragma unroll
      for (int ch = 0; ch < 2; ch++) {
        AFrag b;
        b.u[0] = *(const uint4*)&Ks[c * 16 + lm][ch * 32 + 16 * half];
        b.u[1] = *(const uint4*)&Ks[c * 16 + lm][ch * 32 + 16 * half + 8];
        s[c].v = __builtin_amdgcn_wmma_f32_16x16x32_f16(false, aq[ch].v, false, b.v,
                                                        (short)0, s[c].v, false, false);
      }
    }

    // mask + online softmax (C layout: row = r + 8*half, col = lm)
    int bk[2];
#pragma unroll
    for (int c = 0; c < 2; c++) bk[c] = batch[kb + c * 16 + lm];
    float sv[2][8];
#pragma unroll
    for (int c = 0; c < 2; c++)
#pragma unroll
      for (int r = 0; r < 8; r++)
        sv[c][r] = (bk[c] == bq[r]) ? s[c].f[r] * scale : -INF;

#pragma unroll
    for (int r = 0; r < 8; r++) {
      float mx = fmaxf(sv[0][r], sv[1][r]);
#pragma unroll
      for (int m = 8; m >= 1; m >>= 1) mx = fmaxf(mx, __shfl_xor(mx, m, 16));
      float mnew = fmaxf(mrow[r], mx);
      float alpha = (mnew == -INF) ? 1.0f : __expf(mrow[r] - mnew);
      float ps = 0.0f;
#pragma unroll
      for (int c = 0; c < 2; c++) {
        float e = (sv[c][r] == -INF) ? 0.0f : __expf(sv[c][r] - mnew);
        sv[c][r] = e; ps += e;
      }
#pragma unroll
      for (int m = 8; m >= 1; m >>= 1) ps += __shfl_xor(ps, m, 16);
      lrow[r] = lrow[r] * alpha + ps;
      mrow[r] = mnew;
#pragma unroll
      for (int t = 0; t < 4; t++) out[t].f[r] *= alpha;
      Ps[r + 8 * half][lm]      = (_Float16)sv[0][r];
      Ps[r + 8 * half][16 + lm] = (_Float16)sv[1][r];
    }
    __syncthreads();

    // out += P(16x32) @ V(32x64); V fragments contiguous in dim-major Vt
    AFrag pa;
    pa.u[0] = *(const uint4*)&Ps[lm][16 * half];
    pa.u[1] = *(const uint4*)&Ps[lm][16 * half + 8];
#pragma unroll
    for (int t = 0; t < 4; t++) {
      AFrag b;
      b.u[0] = *(const uint4*)&Vt[t * 16 + lm][16 * half];
      b.u[1] = *(const uint4*)&Vt[t * 16 + lm][16 * half + 8];
      out[t].v = __builtin_amdgcn_wmma_f32_16x16x32_f16(false, pa.v, false, b.v,
                                                        (short)0, out[t].v, false, false);
    }
    __syncthreads();
  }

#pragma unroll
  for (int t = 0; t < 4; t++)
#pragma unroll
    for (int r = 0; r < 8; r++) {
      int row = q0 + r + 8 * half;
      float denom = lrow[r];
      float v = (denom > 0.0f) ? out[t].f[r] / denom : 0.0f;
      attended[(size_t)row * D + colbase + t * 16 + lm] = v;
    }
}

// per-graph segment softmax of w = 0.6*vuln + 0.3*ctx + 0.1*(1/N), temp 5
// (probs.mean(axis=(1,2)) == 1/N exactly: each softmax row sums to 1)
__global__ void seg_softmax_kernel(const float* __restrict__ vuln, const float* __restrict__ ctx,
                                   const int* __restrict__ gstart, float* __restrict__ att) {
  __shared__ float red[256];
  const int g = blockIdx.x, tid = threadIdx.x;
  const int s = gstart[g], e = gstart[g + 1];
  const float mha = 0.1f / (float)N;
  float mx = -__builtin_inff();
  for (int r = s + tid; r < e; r += 256) {
    float w = (0.6f * vuln[r] + 0.3f * ctx[r] + mha) * 5.0f;
    mx = fmaxf(mx, w);
  }
  red[tid] = mx; __syncthreads();
  for (int o = 128; o >= 1; o >>= 1) { if (tid < o) red[tid] = fmaxf(red[tid], red[tid + o]); __syncthreads(); }
  mx = red[0]; __syncthreads();
  float sum = 0.0f;
  for (int r = s + tid; r < e; r += 256) {
    float w = (0.6f * vuln[r] + 0.3f * ctx[r] + mha) * 5.0f;
    sum += __expf(w - mx);
  }
  red[tid] = sum; __syncthreads();
  for (int o = 128; o >= 1; o >>= 1) { if (tid < o) red[tid] += red[tid + o]; __syncthreads(); }
  sum = red[0]; __syncthreads();
  for (int r = s + tid; r < e; r += 256) {
    float w = (0.6f * vuln[r] + 0.3f * ctx[r] + mha) * 5.0f;
    att[r] = __expf(w - mx) / sum;
  }
}

// out = LayerNorm(attended*att + x) * gamma + beta
__global__ void final_ln_kernel(const float* __restrict__ attended, const float* __restrict__ att,
                                const float* __restrict__ x, const float* __restrict__ gamma,
                                const float* __restrict__ beta, float* __restrict__ out) {
  __shared__ float red[256];
  const int n = blockIdx.x, d = threadIdx.x;
  float o = attended[(size_t)n * D + d] * att[n] + x[(size_t)n * D + d];
  red[d] = o; __syncthreads();
  for (int off = 128; off >= 1; off >>= 1) { if (d < off) red[d] += red[d + off]; __syncthreads(); }
  float mu = red[0] / (float)D; __syncthreads();
  float c = o - mu;
  red[d] = c * c; __syncthreads();
  for (int off = 128; off >= 1; off >>= 1) { if (d < off) red[d] += red[d + off]; __syncthreads(); }
  float var = red[0] / (float)D;
  out[(size_t)n * D + d] = c * rsqrtf(var + LN_EPS) * gamma[d] + beta[d];
}

static inline size_t align256(size_t x) { return (x + 255) & ~(size_t)255; }

extern "C" void kernel_launch(void* const* d_in, const int* in_sizes, int n_in,
                              void* d_out, int out_size, void* d_ws, size_t ws_size,
                              hipStream_t stream) {
  (void)in_sizes; (void)n_in; (void)out_size; (void)ws_size;
  const float* x    = (const float*)d_in[0];
  const float* Wq   = (const float*)d_in[1];  const float* bq  = (const float*)d_in[2];
  const float* Wk   = (const float*)d_in[3];  const float* bk  = (const float*)d_in[4];
  const float* Wv   = (const float*)d_in[5];  const float* bv  = (const float*)d_in[6];
  const float* Wv1  = (const float*)d_in[7];  const float* bv1 = (const float*)d_in[8];
  const float* Wv2  = (const float*)d_in[9];  const float* bv2 = (const float*)d_in[10];
  const float* Wc1  = (const float*)d_in[11]; const float* bc1 = (const float*)d_in[12];
  const float* Wc2  = (const float*)d_in[13]; const float* bc2 = (const float*)d_in[14];
  const float* gamma = (const float*)d_in[15];
  const float* beta  = (const float*)d_in[16];
  const int*   batch = (const int*)d_in[17];

  float* outF = (float*)d_out;          // [N*D] normalized output
  float* att  = outF + (size_t)N * D;   // [N]   per-node attention weights

  char* p = (char*)d_ws;
  auto carve = [&](size_t bytes) { char* r = p; p += align256(bytes); return r; };
  _Float16* xh    = (_Float16*)carve((size_t)N * D * 2);
  _Float16* WqT   = (_Float16*)carve((size_t)D * D * 2);        // [D(n)][D(k)]
  _Float16* WkT   = (_Float16*)carve((size_t)D * D * 2);
  _Float16* WvT   = (_Float16*)carve((size_t)D * D * 2);
  _Float16* Wv1T  = (_Float16*)carve((size_t)(D / 2) * D * 2);  // [128][256]
  _Float16* Wc1T  = (_Float16*)carve((size_t)D * D * 2);        // top half, transposed
  _Float16* Qh    = (_Float16*)carve((size_t)N * D * 2);
  _Float16* Kh    = (_Float16*)carve((size_t)N * D * 2);
  _Float16* VhT   = (_Float16*)carve((size_t)D * N * 2);        // [D][N]
  float* h1       = (float*)carve((size_t)N * (D / 2) * 4);
  float* tbuf     = (float*)carve((size_t)N * D * 4);
  float* attended = (float*)carve((size_t)N * D * 4);
  float* gctx     = (float*)carve((size_t)G * D * 4);
  float* gpart    = (float*)carve((size_t)G * D * 4);
  float* vuln     = (float*)carve((size_t)N * 4);
  float* ctxv     = (float*)carve((size_t)N * 4);
  int*   gstart   = (int*)carve((size_t)(G + 1) * 4);

  f32_to_f16_kernel<<<(N * D + 255) / 256, 256, 0, stream>>>(x, xh, N * D);
  f32_to_f16_T_kernel<<<(D * D + 255) / 256, 256, 0, stream>>>(Wq, WqT, D, D);
  f32_to_f16_T_kernel<<<(D * D + 255) / 256, 256, 0, stream>>>(Wk, WkT, D, D);
  f32_to_f16_T_kernel<<<(D * D + 255) / 256, 256, 0, stream>>>(Wv, WvT, D, D);
  f32_to_f16_T_kernel<<<(D * D / 2 + 255) / 256, 256, 0, stream>>>(Wv1, Wv1T, D, D / 2);
  f32_to_f16_T_kernel<<<(D * D + 255) / 256, 256, 0, stream>>>(Wc1, Wc1T, D, D); // top D rows

  find_gstart_kernel<<<1, 32, 0, stream>>>(batch, gstart);

  dim3 blk(32);
  dim3 gFull(N / 16, D / 64);
  dim3 gHalf(N / 16, (D / 2) / 64);
  // Q/K row-major f16; V written transposed [D][N] for the P@V operand
  wmma_gemm_kernel<0, false, false, 1><<<gFull, blk, 0, stream>>>(xh, WqT, bq, nullptr, batch, nullptr, Qh, N, D, D);
  wmma_gemm_kernel<0, false, false, 1><<<gFull, blk, 0, stream>>>(xh, WkT, bk, nullptr, batch, nullptr, Kh, N, D, D);
  wmma_gemm_kernel<0, false, false, 2><<<gFull, blk, 0, stream>>>(xh, WvT, bv, nullptr, batch, nullptr, VhT, N, D, D);

  // vuln MLP layer 1: h1 = relu(x @ Wv1 + bv1)
  wmma_gemm_kernel<1, false, true, 0><<<gHalf, blk, 0, stream>>>(xh, Wv1T, bv1, nullptr, batch, h1, nullptr, N, D, D / 2);

  // graph context
  seg_max_x_kernel<<<G, D, 0, stream>>>(x, gstart, gctx);
  gpart_kernel<<<G, D, 0, stream>>>(gctx, Wc1, gpart);

  // t = tanh(x @ Wc1_top + gpart[batch] + bc1)
  wmma_gemm_kernel<2, true, true, 0><<<gFull, blk, 0, stream>>>(xh, Wc1T, bc1, gpart, batch, tbuf, nullptr, N, D, D);

  // sigmoid heads
  rowdot_sigmoid_kernel<<<N, 32, 0, stream>>>(h1, Wv2, bv2, vuln, D / 2);
  rowdot_sigmoid_kernel<<<N, 32, 0, stream>>>(tbuf, Wc2, bc2, ctxv, D);

  // flash attention
  dim3 gAtt(N / 16, H);
  flash_attn_kernel<<<gAtt, blk, 0, stream>>>(Qh, Kh, VhT, batch, gstart, attended);

  // segment softmax -> att (d_out tail)
  seg_softmax_kernel<<<G, 256, 0, stream>>>(vuln, ctxv, gstart, att);

  // residual + LayerNorm -> d_out
  final_ln_kernel<<<N, 256, 0, stream>>>(attended, att, x, gamma, beta, outF);
}
